// global_memory_lstm_57999238365702
// MI455X (gfx1250) — compile-verified, matrix operand order checked by reading, not run on hardware
//
#include <hip/hip_runtime.h>
#include <hip/hip_bf16.h>

typedef __attribute__((ext_vector_type(16))) __bf16 v16bf;
typedef __attribute__((ext_vector_type(8)))  float  v8f;
typedef __attribute__((ext_vector_type(2)))  float  f32x2;
typedef __attribute__((ext_vector_type(2)))  __bf16 bf16x2;

#define MT 128
#define NT 64
#define KT 32
#define RG 32   // persistent workgroups for recurrence
#define RT 256  // threads per recurrence workgroup

__device__ __forceinline__ unsigned cvt2(float x, float y) {
  union { bf16x2 b; unsigned u; } c;
  c.b = __builtin_convertvector((f32x2){x, y}, bf16x2);  // v_cvt_pk_bf16_f32
  return c.u;
}
__device__ __forceinline__ unsigned short f2bf(float f) {
  union { bf16x2 b; unsigned u; } c;
  c.b = __builtin_convertvector((f32x2){f, 0.f}, bf16x2);
  return (unsigned short)(c.u & 0xFFFFu);
}

// gfx1250 async global->LDS copy (8B per lane), tracked by ASYNCcnt.
__device__ __forceinline__ void async_ld_b64(unsigned lds, unsigned long long gaddr) {
  asm volatile("global_load_async_to_lds_b64 %0, %1, off"
               :: "v"(lds), "v"(gaddr) : "memory");
}

// ---------------------------------------------------------------------------
// out[M,N] = A @ W^T + bias.  A:[M,K] bf16, W:[N,K] bf16 (both ld = K).
// Double-buffered LDS tiles filled by async-to-LDS DMA, consumed by
// v_wmma_f32_16x16x32_bf16.  Block tile 128x64, 8 waves, 32x32 per wave.
// ---------------------------------------------------------------------------
__global__ __launch_bounds__(256) void wmma_gemm_bias_bf16(
    const unsigned short* __restrict__ A, const unsigned short* __restrict__ W,
    const float* __restrict__ bias, float* __restrict__ out,
    int M, int N, int K)
{
  __shared__ unsigned short At[2][MT][36];  // 72B row stride: 8B aligned, bank-spread
  __shared__ unsigned short Bt[2][NT][36];

  const int tid  = threadIdx.x;
  const int lane = tid & 31, wave = tid >> 5;
  const int wm = wave & 3, wn = wave >> 2;       // 4x2 wave grid
  const int half = lane >> 4, l16 = lane & 15;
  const int mBase = blockIdx.x * MT;
  const int nBase = blockIdx.y * NT;

  v8f acc[2][2] = {};
  const int nIter = K / KT;

  // issue async copies for k-tile `kt` into LDS buffer `buf`
  auto issue = [&](int kt, int buf) {
    const int k0 = kt * KT;
    #pragma unroll
    for (int s = 0; s < 4; ++s) {              // A: 128x32 bf16 = 1024 b64 chunks
      int c = tid + s * 256;
      int row = c >> 3, kce = (c & 7) * 4;
      unsigned lds = (unsigned)(unsigned long long)&At[buf][row][kce];
      unsigned long long g = (unsigned long long)(const void*)
          (A + (size_t)(mBase + row) * K + k0 + kce);
      async_ld_b64(lds, g);
    }
    #pragma unroll
    for (int s = 0; s < 2; ++s) {              // B: 64x32 bf16 = 512 b64 chunks
      int c = tid + s * 256;
      int row = c >> 3, kce = (c & 7) * 4;
      unsigned lds = (unsigned)(unsigned long long)&Bt[buf][row][kce];
      unsigned long long g = (unsigned long long)(const void*)
          (W + (size_t)(nBase + row) * K + k0 + kce);
      async_ld_b64(lds, g);
    }
  };

  issue(0, 0);                                  // prologue: tile 0 -> buf 0

  for (int i = 0; i < nIter; ++i) {
    const int buf = i & 1;
    if (i + 1 < nIter) {
      issue(i + 1, buf ^ 1);                    // prefetch next tile
      asm volatile("s_wait_asynccnt 0x6" ::: "memory");  // tile i complete
    } else {
      asm volatile("s_wait_asynccnt 0x0" ::: "memory");
    }
    __syncthreads();

    // gather fragments per ISA 16-bit layout: lane -> M/N = l16, half = lane/16,
    // element pair p -> k = (p<4) ? 8*half+2p : 16+8*half+2(p-4)
    union Frag { v16bf v; unsigned u[8]; };
    Frag a[2], b[2];
    #pragma unroll
    for (int t = 0; t < 2; ++t) {
      int row = wm * 32 + t * 16 + l16;
      int col = wn * 32 + t * 16 + l16;
      #pragma unroll
      for (int p = 0; p < 8; ++p) {
        int kb = (p < 4) ? (8 * half + 2 * p) : (16 + 8 * half + 2 * (p - 4));
        a[t].u[p] = *(const unsigned*)&At[buf][row][kb];
        b[t].u[p] = *(const unsigned*)&Bt[buf][col][kb];
      }
    }
    #pragma unroll
    for (int t = 0; t < 2; ++t)
      #pragma unroll
      for (int t2 = 0; t2 < 2; ++t2)
        acc[t][t2] = __builtin_amdgcn_wmma_f32_16x16x32_bf16(
            false, a[t].v, false, b[t2].v, (short)0, acc[t][t2], false, false);
    __syncthreads();                            // protect buf before overwrite
  }

  // epilogue: C/D layout — lane gives N=l16, VGPR r gives M = r + 8*half
  #pragma unroll
  for (int t = 0; t < 2; ++t) {
    #pragma unroll
    for (int t2 = 0; t2 < 2; ++t2) {
      int col = nBase + wn * 32 + t2 * 16 + l16;
      float bv = bias[col];
      #pragma unroll
      for (int r = 0; r < 8; ++r) {
        int row = mBase + wm * 32 + t * 16 + half * 8 + r;
        out[(size_t)row * N + col] = acc[t][t2][r] + bv;
      }
    }
  }
}

// ---------------------------------------------------------------------------
// Persistent LSTM recurrence: RG resident WGs, one device barrier per step.
// WG g owns h-lanes [g*32, g*32+32): 128 dotps (4 gates x 32 j) of length H.
// Emits hs in bf16 so the output GEMM consumes it directly.
// ---------------------------------------------------------------------------
__global__ __launch_bounds__(RT) void lstm_recurrence(
    const float* __restrict__ x_gates,   // [T, 4H] fp32
    const float* __restrict__ W_hh,      // [4H, H] fp32 (L2-resident)
    const float* __restrict__ b_hh,      // [4H]
    const float* __restrict__ c0,        // [H]
    float* __restrict__ hbuf,            // [2, H] double buffer
    unsigned short* __restrict__ hs_bf,  // [T, H] bf16
    unsigned* __restrict__ bar,
    int T, int H)
{
  __shared__ float h_s[1024];
  __shared__ float red[RT];
  __shared__ float dots[128];
  __shared__ float c_s[32];
  __shared__ float bb[128];

  const int tid   = threadIdx.x;
  const int jBase = blockIdx.x * 32;
  const int r     = tid >> 1;      // which of the 128 dot products
  const int khalf = tid & 1;       // which K-half (512 each)
  const int gate  = r >> 5;        // 0..3 (i,f,g,o)
  const int jj    = r & 31;
  const float* wrow = W_hh + (size_t)(gate * H + jBase + jj) * H + khalf * (H / 2);

  if (tid < 32)  c_s[tid] = c0[jBase + tid];
  if (tid < 128) bb[tid]  = b_hh[(tid >> 5) * H + jBase + (tid & 31)];
  __syncthreads();

  for (int t = 0; t < T; ++t) {
    const float* hr = hbuf + (t & 1) * H;      // h(t-1), visible after barrier
    #pragma unroll
    for (int i = 0; i < 4; ++i) h_s[tid + i * RT] = hr[tid + i * RT];
    __syncthreads();

    float sum = 0.f;
    const float* hh = h_s + khalf * (H / 2);
    #pragma unroll 4
    for (int k = 0; k < H / 2; k += 4) {
      float4 wv = *(const float4*)(wrow + k);
      sum += wv.x * hh[k] + wv.y * hh[k + 1] + wv.z * hh[k + 2] + wv.w * hh[k + 3];
    }
    red[tid] = sum;
    __syncthreads();
    if (khalf == 0) dots[r] = red[2 * r] + red[2 * r + 1];
    __syncthreads();

    if (tid < 32) {
      int j = jBase + tid;
      const float* xg = x_gates + (size_t)t * 4 * H;
      float gi = dots[tid]      + bb[tid]      + xg[j];
      float gf = dots[32 + tid] + bb[32 + tid] + xg[H + j];
      float gg = dots[64 + tid] + bb[64 + tid] + xg[2 * H + j];
      float go = dots[96 + tid] + bb[96 + tid] + xg[3 * H + j];
      float i_ = 1.f / (1.f + __expf(-gi));
      float f_ = 1.f / (1.f + __expf(-gf));
      float g_ = tanhf(gg);
      float o_ = 1.f / (1.f + __expf(-go));
      float cc = f_ * c_s[tid] + i_ * g_;
      c_s[tid] = cc;
      float hn = o_ * tanhf(cc);
      hbuf[((t + 1) & 1) * H + j] = hn;
      hs_bf[(size_t)t * H + j] = f2bf(hn);
    }
    __threadfence();
    __syncthreads();
    if (tid == 0) {
      atomicAdd(bar, 1u);
      unsigned target = (unsigned)(RG * (t + 1));
      volatile unsigned* vb = (volatile unsigned*)bar;
      while (*vb < target) { __builtin_amdgcn_s_sleep(1); }
    }
    __syncthreads();
    __threadfence();
  }
}

// ---------------------------------------------------------------------------
// One-shot fp32 -> bf16 pre-conversion (amortizes conversion out of GEMM loop,
// halves GEMM operand traffic).  8 elements / thread, packed uint4 stores.
// ---------------------------------------------------------------------------
__global__ void cvt_bf16(const float* __restrict__ in, unsigned short* __restrict__ out,
                         long long n)
{
  long long e = ((long long)blockIdx.x * blockDim.x + threadIdx.x) * 8;
  if (e >= n) return;
  float4 a = *(const float4*)(in + e);
  float4 b = *(const float4*)(in + e + 4);
  uint4 o;
  o.x = cvt2(a.x, a.y); o.y = cvt2(a.z, a.w);
  o.z = cvt2(b.x, b.y); o.w = cvt2(b.z, b.w);
  *(uint4*)(out + e) = o;
}

// concat(src,dst) along the channel dim -> bf16 [T, 2C]
__global__ void cvt_concat_bf16(const float* __restrict__ src,
                                const float* __restrict__ dst,
                                unsigned short* __restrict__ out, int T, int C)
{
  long long e = ((long long)blockIdx.x * blockDim.x + threadIdx.x) * 8;
  if (e >= (long long)T * 2 * C) return;
  int row = (int)(e / (2 * C));
  int col = (int)(e % (2 * C));
  const float* s = (col < C) ? (src + (size_t)row * C + col)
                             : (dst + (size_t)row * C + col - C);
  float4 a = *(const float4*)s;
  float4 b = *(const float4*)(s + 4);
  uint4 o;
  o.x = cvt2(a.x, a.y); o.y = cvt2(a.z, a.w);
  o.z = cvt2(b.x, b.y); o.w = cvt2(b.z, b.w);
  *(uint4*)(out + e) = o;
}

__global__ void init_state(const float* __restrict__ h0, float* __restrict__ hbuf,
                           unsigned* __restrict__ bar, int H)
{
  int i = blockIdx.x * blockDim.x + threadIdx.x;
  if (i < H) { hbuf[i] = h0[i]; hbuf[H + i] = 0.f; }
  if (i == 0) *bar = 0u;
}

extern "C" void kernel_launch(void* const* d_in, const int* in_sizes, int n_in,
                              void* d_out, int out_size, void* d_ws, size_t ws_size,
                              hipStream_t stream) {
  (void)in_sizes; (void)n_in; (void)out_size; (void)ws_size;
  const float* src_e = (const float*)d_in[0];
  const float* dst_e = (const float*)d_in[1];
  const float* h0    = (const float*)d_in[2];
  const float* c0    = (const float*)d_in[3];
  const float* W_ih  = (const float*)d_in[4];
  const float* W_hh  = (const float*)d_in[5];
  const float* b_ih  = (const float*)d_in[6];
  const float* b_hh  = (const float*)d_in[7];
  const float* W_lin = (const float*)d_in[8];
  const float* b_lin = (const float*)d_in[9];

  const int T = 8192, IN2 = 1024, H = 1024, H4 = 4096, E = 1024;

  char* p = (char*)d_ws;
  float* x_gates = (float*)p;                 p += (size_t)T * H4 * 4;   // 128 MB
  unsigned short* hs_bf   = (unsigned short*)p; p += (size_t)T * H * 2;  //  16 MB
  unsigned short* A_bf    = (unsigned short*)p; p += (size_t)T * IN2 * 2;//  16 MB
  unsigned short* Wih_bf  = (unsigned short*)p; p += (size_t)H4 * IN2 * 2;//  8 MB
  unsigned short* Wlin_bf = (unsigned short*)p; p += (size_t)E * H * 2;  //   2 MB
  float* hbuf = (float*)p;                    p += (size_t)2 * H * 4;
  unsigned* bar = (unsigned*)p;

  // pre-conversion + state init
  long long nA = (long long)T * IN2, nWih = (long long)H4 * IN2, nWl = (long long)E * H;
  cvt_concat_bf16<<<(unsigned)((nA / 8 + 255) / 256), 256, 0, stream>>>(
      src_e, dst_e, A_bf, T, IN2 / 2);
  cvt_bf16<<<(unsigned)((nWih / 8 + 255) / 256), 256, 0, stream>>>(W_ih, Wih_bf, nWih);
  cvt_bf16<<<(unsigned)((nWl / 8 + 255) / 256), 256, 0, stream>>>(W_lin, Wlin_bf, nWl);
  init_state<<<(H + 255) / 256, 256, 0, stream>>>(h0, hbuf, bar, H);

  // input-side GEMM: x_gates[T,4H] = A_bf @ W_ih^T + b_ih
  dim3 g1(T / MT, H4 / NT);
  wmma_gemm_bias_bf16<<<g1, 256, 0, stream>>>(A_bf, Wih_bf, b_ih, x_gates, T, H4, IN2);

  // sequential recurrence
  lstm_recurrence<<<RG, RT, 0, stream>>>(x_gates, W_hh, b_hh, c0, hbuf, hs_bf, bar, T, H);

  // output GEMM: out[T,E] = hs @ W_lin^T + b_lin
  dim3 g3(T / MT, E / NT);
  wmma_gemm_bias_bf16<<<g3, 256, 0, stream>>>(hs_bf, Wlin_bf, b_lin, (float*)d_out, T, E, H);
}
